// Attention_76802605187548
// MI455X (gfx1250) — compile-verified, matrix operand order checked by reading, not run on hardware
//
#include <hip/hip_runtime.h>
#include <math.h>

// CDNA5 (gfx1250) wave32 WMMA types
typedef __attribute__((ext_vector_type(16))) _Float16 v16h;
typedef __attribute__((ext_vector_type(8)))  float    v8f;

#define CDIV(a, b) (((a) + (b) - 1) / (b))

__device__ inline unsigned pack2h(_Float16 a, _Float16 b) {
    union { _Float16 h[2]; unsigned u; } c;
    c.h[0] = a; c.h[1] = b;
    return c.u;
}

// Build a 16x32 f16 WMMA fragment from an LDS row of 16 dwords (k-pairs).
// ISA 7.12.2 layout: VGPR v holds k pair base (v<4 ? 2v : 16+2(v-4)) + 8*half,
// i.e. dword index (v<4 ? v : v+4) + 4*half.
__device__ inline v16h frag16(const unsigned* rowbase, int half) {
    union { unsigned u[8]; v16h h; } f;
    #pragma unroll
    for (int v = 0; v < 8; ++v) {
        const int dj = (v < 4 ? v : v + 4) + 4 * half;
        f.u[v] = rowbase[dj];
    }
    return f.h;
}

// --------------------------------------------------------------------------
// Tiled implicit-GEMM convolution.
//   Block = 128 threads (4 waves), output tile 64(co) x 128(pixels).
//   K chunked by 32. A (f16 weights, K padded to Kpad%32==0) is staged into
//   LDS with gfx1250 GLOBAL_LOAD_ASYNC_TO_LDS_B128 (ASYNCcnt-tracked, no
//   VGPR round trip); im2col'd B is gathered/packed by the VALU in parallel.
//   Each wave: 4 A frags x 2 B frags -> 8 WMMAs per chunk (32/block-chunk).
// --------------------------------------------------------------------------
__global__ __launch_bounds__(128)
void conv_wmma(const _Float16* __restrict__ in16, const _Float16* __restrict__ wh,
               const float* __restrict__ bias, float* __restrict__ out,
               int CI, int H, int W, int CO, int OH, int OW,
               int KH, int KW, int PAD, int Kpad, int relu)
{
    __shared__ alignas(16) unsigned lA[64 * 16];    // 64 co rows x 32 f16 k
    __shared__ alignas(16) unsigned lB[128 * 16];   // 128 pixels x 32 f16 k
    __shared__ int s_ci[32], s_dy[32], s_dx[32];

    const int tid  = threadIdx.x;
    const int wv   = tid >> 5;
    const int lane = tid & 31;
    const int half = lane >> 4;
    const int idx  = lane & 15;
    const int co0  = blockIdx.x * 64;
    const int p0   = blockIdx.y * 128;
    const int b    = blockIdx.z;
    const int OHW  = OH * OW;
    const int KHW  = KH * KW;

    // raw LDS byte offset of lA (flat shared address truncates to LDS offset)
    const unsigned lAoff = (unsigned)(unsigned long long)(void*)&lA[0];

    v8f acc[4][2] = {};

    for (int k0 = 0; k0 < Kpad; k0 += 32) {
        __syncthreads();
        // k -> (ci, dy, dx); ci = -1 marks zero-padding (k >= K)
        if (tid < 32) {
            const int k = k0 + tid;
            int ci = k / KHW;
            const int r = k - ci * KHW;
            if (ci >= CI) ci = -1;
            s_ci[tid] = ci;
            s_dy[tid] = r / KW;
            s_dx[tid] = r - (r / KW) * KW;
        }
        // stage A asynchronously: 64 rows x 64B = 256 x b128 transfers
        #pragma unroll
        for (int i = tid; i < 256; i += 128) {
            const int m = i >> 2, d = i & 3;               // 4 x 16B per row
            const _Float16* g = wh + (co0 + m) * Kpad + k0 + d * 8;
            const unsigned lo = lAoff + (unsigned)i * 16u;
            asm volatile("global_load_async_to_lds_b128 %0, %1, off"
                         :: "v"(lo), "v"(g) : "memory");
        }
        __syncthreads();   // publish s_ci/s_dy/s_dx for the B gather

        // stage B: im2col gather from f16 activations, packed 2/dword
        #pragma unroll
        for (int i = tid; i < 2048; i += 128) {
            const int n = i >> 4, d = i & 15;
            const int p = p0 + n;
            _Float16 h0 = (_Float16)0.f, h1 = (_Float16)0.f;
            if (p < OHW) {
                const int oy = p / OW, ox = p - (p / OW) * OW;
                #pragma unroll
                for (int t = 0; t < 2; ++t) {
                    const int kr = 2 * d + t;
                    const int ci = s_ci[kr];
                    if (ci >= 0) {
                        const int iy = oy + s_dy[kr] - PAD;
                        const int ix = ox + s_dx[kr] - PAD;
                        if (iy >= 0 && iy < H && ix >= 0 && ix < W) {
                            const _Float16 v = in16[((b * CI + ci) * H + iy) * W + ix];
                            if (t == 0) h0 = v; else h1 = v;
                        }
                    }
                }
            }
            lB[i] = pack2h(h0, h1);
        }
        // wait for this wave's async LDS copies, then barrier: all staged
        asm volatile("s_wait_asynccnt 0x0" ::: "memory");
        __syncthreads();

        // compute: 2 B fragments, 4 A fragments, 8 WMMAs per wave
        const v16h bv0 = frag16(&lB[(wv * 32 + idx) * 16], half);
        const v16h bv1 = frag16(&lB[(wv * 32 + 16 + idx) * 16], half);
        #pragma unroll
        for (int mt = 0; mt < 4; ++mt) {
            const v16h av = frag16(&lA[(mt * 16 + idx) * 16], half);
            acc[mt][0] = __builtin_amdgcn_wmma_f32_16x16x32_f16(
                false, av, false, bv0, (short)0, acc[mt][0], false, false);
            acc[mt][1] = __builtin_amdgcn_wmma_f32_16x16x32_f16(
                false, av, false, bv1, (short)0, acc[mt][1], false, false);
        }
    }

    // store (C layout: VGPR r -> M = r + 8*half, lane&15 -> N)
    #pragma unroll
    for (int nt = 0; nt < 2; ++nt) {
        const int p = p0 + wv * 32 + nt * 16 + idx;
        if (p < OHW) {
            #pragma unroll
            for (int mt = 0; mt < 4; ++mt) {
                #pragma unroll
                for (int r = 0; r < 8; ++r) {
                    const int mo = co0 + mt * 16 + r + 8 * half;
                    float v = acc[mt][nt][r] + bias[mo];
                    if (relu) v = fmaxf(v, 0.0f);
                    out[(b * CO + mo) * OHW + p] = v;
                }
            }
        }
    }
}

// --------------------------------------------------------------------------
// WMMA GEMM: C[z] (MxN) = A[z] (MxK, row major, f32) * Bw^T (Bw NxK f32).
// One wave per 16x16 tile (used once: src_feat = src @ Wi^T, 1.9 GFLOP).
// --------------------------------------------------------------------------
__global__ __launch_bounds__(32)
void gemm_wmma_bt(const float* __restrict__ A, const float* __restrict__ Bw,
                  float* __restrict__ C, int M, int N, int K,
                  long strideA, long strideC)
{
    const int lane = threadIdx.x & 31;
    const int half = lane >> 4;
    const int idx  = lane & 15;
    const int m0   = blockIdx.x * 16;
    const int n0   = blockIdx.y * 16;
    const float* Ab = A + (long)blockIdx.z * strideA;
    float*       Cb = C + (long)blockIdx.z * strideC;
    const int mrow = m0 + idx;
    const int ncol = n0 + idx;

    v8f acc = {};
    for (int k0 = 0; k0 < K; k0 += 32) {
        v16h av, bv;
        #pragma unroll
        for (int v = 0; v < 8; ++v) {
            const int kb = (v < 4 ? 2 * v : 16 + 2 * (v - 4)) + 8 * half;
            #pragma unroll
            for (int t = 0; t < 2; ++t) {
                const int kk = k0 + kb + t;
                float aval = (mrow < M && kk < K) ? Ab[(long)mrow * K + kk] : 0.0f;
                float bval = (ncol < N && kk < K) ? Bw[(long)ncol * K + kk] : 0.0f;
                av[2 * v + t] = (_Float16)aval;
                bv[2 * v + t] = (_Float16)bval;
            }
        }
        acc = __builtin_amdgcn_wmma_f32_16x16x32_f16(
            false, av, false, bv, (short)0, acc, false, false);
    }
    if (ncol < N) {
        #pragma unroll
        for (int r = 0; r < 8; ++r) {
            const int mo = m0 + r + 8 * half;
            if (mo < M) Cb[(long)mo * N + ncol] = acc[r];
        }
    }
}

// --------------------------------------------------------------------------
// Conversions
// --------------------------------------------------------------------------
__global__ void cvt_f32_to_f16(const float* __restrict__ in, _Float16* __restrict__ out,
                               long total)
{
    long i = (long)blockIdx.x * blockDim.x + threadIdx.x;
    if (i < total) out[i] = (_Float16)in[i];
}

// weights (CO, K) f32 -> (CO, Kpad) f16, zero padded
__global__ void cvt_weights(const float* __restrict__ w, _Float16* __restrict__ wh,
                            int K, int Kpad, long total)
{
    long i = (long)blockIdx.x * blockDim.x + threadIdx.x;
    if (i >= total) return;
    int k = (int)(i % Kpad);
    long co = i / Kpad;
    wh[i] = (k < K) ? (_Float16)w[co * K + k] : (_Float16)0.f;
}

// --------------------------------------------------------------------------
// Max pool (-inf padding); reads f32, writes f16 (feeds next conv only).
// --------------------------------------------------------------------------
__global__ void maxpool_f16(const float* __restrict__ in, _Float16* __restrict__ out,
                            int C, int H, int W,
                            int kh, int kw, int sh, int sw, int ph, int pw,
                            int OH, int OW, long total)
{
    long i = (long)blockIdx.x * blockDim.x + threadIdx.x;
    if (i >= total) return;
    int ox = (int)(i % OW); long r = i / OW;
    int oy = (int)(r % OH); r /= OH;
    int c  = (int)(r % C);
    int b  = (int)(r / C);
    float m = -3.402823466e38f;
    for (int y = 0; y < kh; ++y)
        for (int x = 0; x < kw; ++x) {
            int iy = oy * sh - ph + y, ix = ox * sw - pw + x;
            if (iy >= 0 && iy < H && ix >= 0 && ix < W)
                m = fmaxf(m, in[(((long)b * C + c) * H + iy) * W + ix]);
        }
    out[i] = (_Float16)m;
}

// --------------------------------------------------------------------------
// BatchNorm (training stats over B,H,W)
// --------------------------------------------------------------------------
__global__ __launch_bounds__(256)
void bn_stats(const float* __restrict__ x, float* __restrict__ stats, int C, int HW)
{
    __shared__ float s1[256], s2[256];
    const int c = blockIdx.x, tid = threadIdx.x;
    const int n = 16 * HW;
    float sum = 0.0f, sq = 0.0f;
    for (int i = tid; i < n; i += 256) {
        int b = i / HW, r = i - b * HW;
        float v = x[((long)b * C + c) * HW + r];
        sum += v; sq += v * v;
    }
    s1[tid] = sum; s2[tid] = sq; __syncthreads();
    for (int o = 128; o > 0; o >>= 1) {
        if (tid < o) { s1[tid] += s1[tid + o]; s2[tid] += s2[tid + o]; }
        __syncthreads();
    }
    if (tid == 0) {
        float m = s1[0] / n;
        stats[c] = m;
        stats[C + c] = s2[0] / n - m * m;
    }
}

// normalize + ReLU: f32 in-place, plus f16 copy for the next conv
__global__ void bn_apply_relu(float* __restrict__ x, _Float16* __restrict__ xh,
                              const float* __restrict__ g, const float* __restrict__ bt,
                              const float* __restrict__ stats,
                              int C, int HW, long total)
{
    long i = (long)blockIdx.x * blockDim.x + threadIdx.x;
    if (i >= total) return;
    int c = (int)((i / HW) % C);
    float m = stats[c], v = stats[C + c];
    float y = (x[i] - m) * rsqrtf(v + 1e-5f) * g[c] + bt[c];
    y = fmaxf(y, 0.0f);
    x[i]  = y;
    xh[i] = (_Float16)y;
}

// (b, c, l) -> (b, l, c)
__global__ void transpose_src(const float* __restrict__ in, float* __restrict__ out,
                              int C, int L, long total)
{
    long i = (long)blockIdx.x * blockDim.x + threadIdx.x;
    if (i >= total) return;
    int c = (int)(i % C); long r = i / C;
    int l = (int)(r % L);
    int b = (int)(r / L);
    out[i] = in[((long)b * C + c) * L + l];
}

// --------------------------------------------------------------------------
// Attention-LSTM decoder scan: one block per batch element, 49 steps.
// --------------------------------------------------------------------------
__global__ __launch_bounds__(128)
void decoder_scan(const float* __restrict__ src, const float* __restrict__ sf,
                  const int* __restrict__ text, const float* __restrict__ emb_tab,
                  const float* __restrict__ Wh, const float* __restrict__ bh,
                  const float* __restrict__ wsv,
                  const float* __restrict__ Wih, const float* __restrict__ bih,
                  const float* __restrict__ Whh, const float* __restrict__ bhh,
                  float* __restrict__ hs, int Lp, int Tsteps)
{
    __shared__ float s_h[128], s_c[128], s_proj[128], s_emb[128];
    __shared__ float s_ctx[512], s_gates[512];
    __shared__ float s_logit[903];
    __shared__ float s_red[128];
    const int tid = threadIdx.x;
    const int b   = blockIdx.x;
    const float* srcb = src + (long)b * Lp * 512;
    const float* sfb  = sf  + (long)b * Lp * 128;

    s_h[tid] = 0.0f; s_c[tid] = 0.0f;
    __syncthreads();

    for (int t = 0; t < Tsteps; ++t) {
        const int tok = text[b * (Tsteps + 1) + t];
        s_emb[tid] = emb_tab[tok * 128 + tid];
        float p = bh[tid];
        for (int k = 0; k < 128; ++k) p += s_h[k] * Wh[tid * 128 + k];
        s_proj[tid] = p;
        __syncthreads();
        for (int l = tid; l < Lp; l += 128) {
            const float* row = sfb + (long)l * 128;
            float s = 0.0f;
            for (int h = 0; h < 128; ++h) s += tanhf(row[h] + s_proj[h]) * wsv[h];
            s_logit[l] = s;
        }
        __syncthreads();
        float mx = -3.402823466e38f;
        for (int l = tid; l < Lp; l += 128) mx = fmaxf(mx, s_logit[l]);
        s_red[tid] = mx; __syncthreads();
        for (int o = 64; o > 0; o >>= 1) {
            if (tid < o) s_red[tid] = fmaxf(s_red[tid], s_red[tid + o]);
            __syncthreads();
        }
        mx = s_red[0]; __syncthreads();
        float sum = 0.0f;
        for (int l = tid; l < Lp; l += 128) {
            float e = __expf(s_logit[l] - mx);
            s_logit[l] = e; sum += e;
        }
        s_red[tid] = sum; __syncthreads();
        for (int o = 64; o > 0; o >>= 1) {
            if (tid < o) s_red[tid] += s_red[tid + o];
            __syncthreads();
        }
        const float inv = 1.0f / s_red[0];
        __syncthreads();
        for (int r = 0; r < 4; ++r) {
            const int c = tid + 128 * r;
            float a = 0.0f;
            for (int l = 0; l < Lp; ++l) a += s_logit[l] * srcb[(long)l * 512 + c];
            s_ctx[c] = a * inv;
        }
        __syncthreads();
        for (int r = 0; r < 4; ++r) {
            const int j = tid + 128 * r;
            const float* wi = Wih + (long)j * 640;
            const float* wh = Whh + (long)j * 128;
            float g = bih[j] + bhh[j];
            for (int k = 0; k < 512; ++k) g += s_ctx[k] * wi[k];
            for (int k = 0; k < 128; ++k) g += s_emb[k] * wi[512 + k];
            for (int k = 0; k < 128; ++k) g += s_h[k] * wh[k];
            s_gates[j] = g;
        }
        __syncthreads();
        const float ig = s_gates[tid], fg = s_gates[128 + tid];
        const float gg = s_gates[256 + tid], og = s_gates[384 + tid];
        const float si  = 1.0f / (1.0f + __expf(-ig));
        const float sfr = 1.0f / (1.0f + __expf(-fg));
        const float so  = 1.0f / (1.0f + __expf(-og));
        const float cn = sfr * s_c[tid] + si * tanhf(gg);
        const float hn = so * tanhf(cn);
        __syncthreads();
        s_c[tid] = cn; s_h[tid] = hn;
        hs[((long)b * Tsteps + t) * 128 + tid] = hn;
        __syncthreads();
    }
}

__global__ void out_proj(const float* __restrict__ hs, const float* __restrict__ Wg,
                         const float* __restrict__ bg, float* __restrict__ out, int total)
{
    int i = blockIdx.x * blockDim.x + threadIdx.x;
    if (i >= total) return;
    int k = i % 245;
    int bt = i / 245;
    const float* h = hs + (long)bt * 128;
    const float* w = Wg + (long)k * 128;
    float acc = bg[k];
    for (int j = 0; j < 128; ++j) acc += h[j] * w[j];
    out[i] = acc;
}

// --------------------------------------------------------------------------
// Orchestration
// --------------------------------------------------------------------------
extern "C" void kernel_launch(void* const* d_in, const int* in_sizes, int n_in,
                              void* d_out, int out_size, void* d_ws, size_t ws_size,
                              hipStream_t stream)
{
    (void)in_sizes; (void)n_in; (void)out_size; (void)ws_size;
    const float* image = (const float*)d_in[0];
    const int*   text  = (const int*)d_in[1];
    const float* cw[7] = { (const float*)d_in[2], (const float*)d_in[4], (const float*)d_in[6],
                           (const float*)d_in[8], (const float*)d_in[10], (const float*)d_in[12],
                           (const float*)d_in[14] };
    const float* cb[7] = { (const float*)d_in[3], (const float*)d_in[5], (const float*)d_in[7],
                           (const float*)d_in[9], (const float*)d_in[11], (const float*)d_in[13],
                           (const float*)d_in[15] };
    const float* g2  = (const float*)d_in[16]; const float* bt2 = (const float*)d_in[17];
    const float* g4  = (const float*)d_in[18]; const float* bt4 = (const float*)d_in[19];
    const float* g6  = (const float*)d_in[20]; const float* bt6 = (const float*)d_in[21];
    const float* emb = (const float*)d_in[22];
    const float* Wi  = (const float*)d_in[23];
    const float* Wh  = (const float*)d_in[24]; const float* bh  = (const float*)d_in[25];
    const float* wsv = (const float*)d_in[26];
    const float* Wih = (const float*)d_in[27]; const float* bih = (const float*)d_in[28];
    const float* Whh = (const float*)d_in[29]; const float* bhh = (const float*)d_in[30];
    const float* Wg  = (const float*)d_in[31]; const float* bg  = (const float*)d_in[32];
    float* outp = (float*)d_out;

    // workspace layout (floats):
    //   F1 [0, 67.2M)            live f32 activation (only one at a time)
    //   F2 [67.2M, 74.7M)        src (b,l,c) f32
    //   H  [74.7M, 83.2M)        f16 activation feeding the next conv
    //   WH [83.2M, 86.2M)        f16 padded weights
    //   stats [86.2M, +1024)
    float* WS   = (float*)d_ws;
    float* F1   = WS;
    float* F2   = WS + 67200000L;
    _Float16* Hb  = (_Float16*)(WS + 74700000L);
    _Float16* WHb = (_Float16*)(WS + 83200000L);
    float* stats = WS + 86200000L;
    float* srcF  = F1;                 // src_feat reuses F1 after encoder
    float* hsBuf = F1 + 2000000L;      // (16,49,128)

    const int  K[7]    = { 9, 576, 1152, 2304, 2304, 4608, 2048 };
    const int  Kpad[7] = { 32, 576, 1152, 2304, 2304, 4608, 2048 };
    const int  COl[7]  = { 64, 128, 256, 256, 512, 512, 512 };
    const long whOff[7] = { 0, 2048, 75776, 370688, 960512, 2140160, 4499456 };
    _Float16* wh[7];
    for (int l = 0; l < 7; ++l) wh[l] = WHb + whOff[l];

    // ---- weight + image conversion ----
    for (int l = 0; l < 7; ++l) {
        long tot = (long)COl[l] * Kpad[l];
        cvt_weights<<<CDIV(tot, 256), 256, 0, stream>>>(cw[l], wh[l], K[l], Kpad[l], tot);
    }
    {   long tot = 16L * 1 * 128 * 512;
        cvt_f32_to_f16<<<CDIV(tot, 256), 256, 0, stream>>>(image, Hb, tot); }

    const dim3 blk(128);

    // ---- encoder ----
    conv_wmma<<<dim3(64 / 64, CDIV(128 * 512, 128), 16), blk, 0, stream>>>(
        Hb, wh[0], cb[0], F1, 1, 128, 512, 64, 128, 512, 3, 3, 1, 32, 1);
    {   long tot = 16L * 64 * 64 * 256;
        maxpool_f16<<<CDIV(tot, 256), 256, 0, stream>>>(
            F1, Hb, 64, 128, 512, 2, 2, 2, 2, 0, 0, 64, 256, tot); }
    conv_wmma<<<dim3(128 / 64, CDIV(64 * 256, 128), 16), blk, 0, stream>>>(
        Hb, wh[1], cb[1], F1, 64, 64, 256, 128, 64, 256, 3, 3, 1, 576, 1);
    {   long tot = 16L * 128 * 32 * 128;
        maxpool_f16<<<CDIV(tot, 256), 256, 0, stream>>>(
            F1, Hb, 128, 64, 256, 2, 2, 2, 2, 0, 0, 32, 128, tot); }
    conv_wmma<<<dim3(256 / 64, CDIV(32 * 128, 128), 16), blk, 0, stream>>>(
        Hb, wh[2], cb[2], F1, 128, 32, 128, 256, 32, 128, 3, 3, 1, 1152, 0);
    bn_stats<<<256, 256, 0, stream>>>(F1, stats, 256, 32 * 128);
    {   long tot = 16L * 256 * 32 * 128;
        bn_apply_relu<<<CDIV(tot, 256), 256, 0, stream>>>(
            F1, Hb, g2, bt2, stats, 256, 32 * 128, tot); }
    conv_wmma<<<dim3(256 / 64, CDIV(32 * 128, 128), 16), blk, 0, stream>>>(
        Hb, wh[3], cb[3], F1, 256, 32, 128, 256, 32, 128, 3, 3, 1, 2304, 1);
    {   long tot = 16L * 256 * 16 * 129;
        maxpool_f16<<<CDIV(tot, 256), 256, 0, stream>>>(
            F1, Hb, 256, 32, 128, 2, 2, 2, 1, 0, 1, 16, 129, tot); }
    conv_wmma<<<dim3(512 / 64, CDIV(16 * 129, 128), 16), blk, 0, stream>>>(
        Hb, wh[4], cb[4], F1, 256, 16, 129, 512, 16, 129, 3, 3, 1, 2304, 0);
    bn_stats<<<512, 256, 0, stream>>>(F1, stats, 512, 16 * 129);
    {   long tot = 16L * 512 * 16 * 129;
        bn_apply_relu<<<CDIV(tot, 256), 256, 0, stream>>>(
            F1, Hb, g4, bt4, stats, 512, 16 * 129, tot); }
    conv_wmma<<<dim3(512 / 64, CDIV(16 * 129, 128), 16), blk, 0, stream>>>(
        Hb, wh[5], cb[5], F1, 512, 16, 129, 512, 16, 129, 3, 3, 1, 4608, 1);
    {   long tot = 16L * 512 * 8 * 130;
        maxpool_f16<<<CDIV(tot, 256), 256, 0, stream>>>(
            F1, Hb, 512, 16, 129, 2, 2, 2, 1, 0, 1, 8, 130, tot); }
    conv_wmma<<<dim3(512 / 64, CDIV(7 * 129, 128), 16), blk, 0, stream>>>(
        Hb, wh[6], cb[6], F1, 512, 8, 130, 512, 7, 129, 2, 2, 0, 2048, 0);
    bn_stats<<<512, 256, 0, stream>>>(F1, stats, 512, 7 * 129);
    {   long tot = 16L * 512 * 7 * 129;
        bn_apply_relu<<<CDIV(tot, 256), 256, 0, stream>>>(
            F1, Hb, g6, bt6, stats, 512, 7 * 129, tot); }

    // ---- decoder ----
    const int L = 7 * 129;   // 903
    {   long tot = 16L * L * 512;
        transpose_src<<<CDIV(tot, 256), 256, 0, stream>>>(F1, F2, 512, L, tot); }
    gemm_wmma_bt<<<dim3(CDIV(L, 16), 128 / 16, 16), dim3(32), 0, stream>>>(
        F2, Wi, srcF, L, 128, 512, (long)L * 512, (long)L * 128);
    decoder_scan<<<16, 128, 0, stream>>>(
        F2, srcF, text, emb, Wh, bh, wsv, Wih, bih, Whh, bhh, hsBuf, L, 49);
    {   int tot = 16 * 49 * 245;
        out_proj<<<CDIV(tot, 256), 256, 0, stream>>>(hsBuf, Wg, bg, outp, tot); }
}
